// SimplifiedCircuitDecoder_58067957842538
// MI455X (gfx1250) — matrix operand order repaired and checked by reading, not compiled, not run on hardware
//
#include <hip/hip_runtime.h>
#include <hip/hip_bf16.h>

typedef __attribute__((ext_vector_type(16))) _Float16 v16h;
typedef __attribute__((ext_vector_type(8)))  _Float16 v8h;
typedef __attribute__((ext_vector_type(8)))  float    v8f;

#define BDIM   2048
#define NNODE  10
#define NKEY   11
#define HDIM   256
#define LDIM   8
#define NLAYER 4
#define FFDIM  1024
#define NHEAD  8
#define HEADD  32

#define ASTR 40   // padded LDS row stride (halves): 80B -> conflict-free fragment reads
#define BSTR 40

// ------------------------------------------------------------------
// zero fill (edge output region relies on it; diagonal stays zero)
// ------------------------------------------------------------------
__global__ void fill_zero_kernel(float* __restrict__ p, int n) {
  int i = blockIdx.x * blockDim.x + threadIdx.x;
  if (i < n) p[i] = 0.0f;
}

// ------------------------------------------------------------------
// fp32 [K,Nc] -> fp16 transposed [Nc,K]  (weight pre-pass, coalesced writes)
// ------------------------------------------------------------------
__global__ void transpose_f16_kernel(const float* __restrict__ W, _Float16* __restrict__ Wt,
                                     int K, int Nc) {
  int idx = blockIdx.x * blockDim.x + threadIdx.x;
  if (idx >= K * Nc) return;
  int n = idx / K, k = idx - n * K;
  Wt[idx] = (_Float16)W[(size_t)k * Nc + n];
}

// ------------------------------------------------------------------
// block-wide LayerNorm over 256 elements (one row per 256-thread block)
// ------------------------------------------------------------------
__device__ __forceinline__ float block_ln256(float x, float g, float be, float* red) {
  int t = threadIdx.x;
  red[t] = x; __syncthreads();
  for (int s = 128; s > 0; s >>= 1) { if (t < s) red[t] += red[t + s]; __syncthreads(); }
  float mean = red[0] * (1.0f / 256.0f);
  __syncthreads();
  float d = x - mean;
  red[t] = d * d; __syncthreads();
  for (int s = 128; s > 0; s >>= 1) { if (t < s) red[t] += red[t + s]; __syncthreads(); }
  float var = red[0] * (1.0f / 256.0f);
  return d * rsqrtf(var + 1e-5f) * g + be;
}

// ------------------------------------------------------------------
// context encoder: Linear(8->256) -> LN -> ReLU -> Linear(256->256) -> LN
// ------------------------------------------------------------------
__global__ __launch_bounds__(256) void ctx_encoder_kernel(
    const float* __restrict__ latent,
    const float* __restrict__ w1, const float* __restrict__ b1,
    const float* __restrict__ g1, const float* __restrict__ be1,
    const float* __restrict__ w2, const float* __restrict__ b2,
    const float* __restrict__ g2, const float* __restrict__ be2,
    float* __restrict__ ctx) {
  __shared__ float xs[LDIM];
  __shared__ float ss[HDIM];
  __shared__ float red[HDIM];
  int b = blockIdx.x, t = threadIdx.x;
  if (t < LDIM) xs[t] = latent[b * LDIM + t];
  __syncthreads();
  float y = b1[t];
  #pragma unroll
  for (int k = 0; k < LDIM; ++k) y += xs[k] * w1[k * HDIM + t];
  y = block_ln256(y, g1[t], be1[t], red);
  y = fmaxf(y, 0.0f);
  ss[t] = y;
  __syncthreads();
  float z = b2[t];
  for (int k = 0; k < HDIM; ++k) z += ss[k] * w2[k * HDIM + t];
  z = block_ln256(z, g2[t], be2[t], red);
  ctx[(size_t)b * HDIM + t] = z;
}

// ------------------------------------------------------------------
// node count head on latent[:, :2]
// ------------------------------------------------------------------
__global__ void node_count_kernel(const float* __restrict__ latent,
                                  const float* __restrict__ w1, const float* __restrict__ b1,
                                  const float* __restrict__ w2, const float* __restrict__ b2,
                                  float* __restrict__ out) {
  int b = blockIdx.x * blockDim.x + threadIdx.x;
  if (b >= BDIM) return;
  float l0 = latent[b * LDIM + 0], l1 = latent[b * LDIM + 1];
  float o0 = b2[0], o1 = b2[1], o2 = b2[2];
  for (int j = 0; j < HDIM / 4; ++j) {
    float hd = fmaxf(l0 * w1[j] + l1 * w1[HDIM / 4 + j] + b1[j], 0.0f);
    o0 += hd * w2[j * 3 + 0];
    o1 += hd * w2[j * 3 + 1];
    o2 += hd * w2[j * 3 + 2];
  }
  out[b * 3 + 0] = o0; out[b * 3 + 1] = o1; out[b * 3 + 2] = o2;
}

// ------------------------------------------------------------------
// build ne, h (queries), mem = concat(ctx, ne)
// ------------------------------------------------------------------
__global__ void build_embed_kernel(const int* __restrict__ types,
                                   const float* __restrict__ type_emb,
                                   const float* __restrict__ pos_emb,
                                   const float* __restrict__ ctx,
                                   float* __restrict__ ne, float* __restrict__ h,
                                   float* __restrict__ mem) {
  int idx = blockIdx.x * blockDim.x + threadIdx.x;
  if (idx >= BDIM * NNODE * HDIM) return;
  int c = idx % HDIM;
  int n = (idx / HDIM) % NNODE;
  int b = idx / (HDIM * NNODE);
  int ty = types[b * NNODE + n];
  float pv = pos_emb[n * HDIM + c];
  float cv = ctx[(size_t)b * HDIM + c];
  float nev = type_emb[ty * HDIM + c] + pv;
  ne[idx] = nev;
  h[idx] = cv + pv;
  mem[((size_t)b * NKEY + n + 1) * HDIM + c] = nev;
  if (n == 0) mem[((size_t)b * NKEY) * HDIM + c] = cv;
}

// ------------------------------------------------------------------
// async copy helpers (CDNA5 GLOBAL_LOAD_ASYNC_TO_LDS, tracked by ASYNCcnt)
// ------------------------------------------------------------------
__device__ __forceinline__ void async_copy_b128(const void* gsrc, void* ldst) {
  unsigned lds = (unsigned)(unsigned long long)ldst;           // low 32b of LDS flat addr
  unsigned long long ga = (unsigned long long)gsrc;
  asm volatile("global_load_async_to_lds_b128 %0, %1, off"
               :: "v"(lds), "v"(ga) : "memory");
}
__device__ __forceinline__ void wait_async() {
  asm volatile("s_wait_asynccnt 0x0" ::: "memory");
}

// ------------------------------------------------------------------
// WMMA GEMM staging: A tile 128x32 (fp32->fp16 via VALU + ds_store_b128),
// B tile 64x32 k-contiguous per column, staged with async-to-LDS b128.
// ------------------------------------------------------------------
__device__ __forceinline__ void stage_A(const float* __restrict__ A, _Float16* As,
                                        int m0, int K, int kt, int tid) {
  #pragma unroll
  for (int p = 0; p < 4; ++p) {
    int q = tid + p * 128;
    int rr = q >> 2, c8 = (q & 3) << 3;
    const float* src = A + (size_t)(m0 + rr) * K + kt + c8;
    float4 lo = *(const float4*)(src);
    float4 hi = *(const float4*)(src + 4);
    v8h d;
    d[0] = (_Float16)lo.x; d[1] = (_Float16)lo.y; d[2] = (_Float16)lo.z; d[3] = (_Float16)lo.w;
    d[4] = (_Float16)hi.x; d[5] = (_Float16)hi.y; d[6] = (_Float16)hi.z; d[7] = (_Float16)hi.w;
    *(v8h*)&As[rr * ASTR + c8] = d;
  }
}

__device__ __forceinline__ void stage_B(const _Float16* __restrict__ Wt, _Float16* Bs,
                                        int n0, int K, int kt, int tid) {
  #pragma unroll
  for (int p = 0; p < 2; ++p) {
    int q = tid + p * 128;
    int cc = q >> 2, k8 = (q & 3) << 3;
    const _Float16* src = Wt + (size_t)(n0 + cc) * K + kt + k8;
    async_copy_b128((const void*)src, (void*)&Bs[cc * BSTR + k8]);
  }
}

// per-wave 32x64 output: 2 A fragments, 4 B fragments (each reused twice) -> 8 WMMA
__device__ __forceinline__ void compute_tile(const _Float16* As, const _Float16* Bs,
                                             int wave, int hf, int r, v8f* acc) {
  v16h a[2];
  #pragma unroll
  for (int af = 0; af < 2; ++af) {
    int arow = (wave * 32 + af * 16 + r) * ASTR + hf * 8;
    v8h a0 = *(const v8h*)&As[arow];
    v8h a1 = *(const v8h*)&As[arow + 16];
    a[af] = __builtin_shufflevector(a0, a1, 0, 1, 2, 3, 4, 5, 6, 7,
                                    8, 9, 10, 11, 12, 13, 14, 15);
  }
  #pragma unroll
  for (int c4 = 0; c4 < 4; ++c4) {
    int bb = (c4 * 16 + r) * BSTR + hf * 8;
    v8h b0 = *(const v8h*)&Bs[bb];
    v8h b1 = *(const v8h*)&Bs[bb + 16];
    v16h b = __builtin_shufflevector(b0, b1, 0, 1, 2, 3, 4, 5, 6, 7,
                                     8, 9, 10, 11, 12, 13, 14, 15);
    acc[c4 * 2 + 0] = __builtin_amdgcn_wmma_f32_16x16x32_f16(false, a[0], false, b,
                                                             (short)0, acc[c4 * 2 + 0], false, false);
    acc[c4 * 2 + 1] = __builtin_amdgcn_wmma_f32_16x16x32_f16(false, a[1], false, b,
                                                             (short)0, acc[c4 * 2 + 1], false, false);
  }
}

// ------------------------------------------------------------------
// WMMA tiled GEMM: C[M,Nc] = A[M,K] @ Wt^T (+bias)(+relu)
// 128 threads = 4 waves; block tile 128x64; double-buffered, K pipelined by 64.
// flags: bit0 = add bias, bit1 = relu. Requires M%128==0, K%64==0, Nc%64==0.
// ------------------------------------------------------------------
__global__ __launch_bounds__(128) void wmma_gemm_kernel(
    const float* __restrict__ A, const _Float16* __restrict__ Wt,
    const float* __restrict__ bias, float* __restrict__ C,
    int M, int K, int Nc, int flags) {
  __shared__ _Float16 As[2][128 * ASTR];
  __shared__ _Float16 Bs[2][64 * BSTR];
  int tid = threadIdx.x;
  int n0 = blockIdx.x * 64;
  int m0 = blockIdx.y * 128;
  int wave = tid >> 5;
  int lane = tid & 31;
  int hf = lane >> 4;
  int r  = lane & 15;
  (void)M;

  v8f acc[8] = {};

  stage_A(A, As[0], m0, K, 0, tid);
  stage_B(Wt, Bs[0], n0, K, 0, tid);
  wait_async();
  __syncthreads();
  for (int kt = 0; kt < K; kt += 64) {
    if (kt + 32 < K) { stage_A(A, As[1], m0, K, kt + 32, tid); stage_B(Wt, Bs[1], n0, K, kt + 32, tid); }
    compute_tile(As[0], Bs[0], wave, hf, r, acc);
    wait_async();
    __syncthreads();
    if (kt + 64 < K) { stage_A(A, As[0], m0, K, kt + 64, tid); stage_B(Wt, Bs[0], n0, K, kt + 64, tid); }
    compute_tile(As[1], Bs[1], wave, hf, r, acc);
    wait_async();
    __syncthreads();
  }

  #pragma unroll
  for (int c4 = 0; c4 < 4; ++c4) {
    #pragma unroll
    for (int af = 0; af < 2; ++af) {
      #pragma unroll
      for (int j = 0; j < 8; ++j) {
        int row = m0 + wave * 32 + af * 16 + hf * 8 + j;  // C/D: lanes 16-31 -> M+8
        int col = n0 + c4 * 16 + r;
        float v = acc[c4 * 2 + af][j];
        if (flags & 1) v += bias[col];
        if (flags & 2) v = fmaxf(v, 0.0f);
        C[(size_t)row * Nc + col] = v;
      }
    }
  }
}

// ------------------------------------------------------------------
// attention: one thread per (query n, head); query n attends keys m <= n
// ------------------------------------------------------------------
__global__ __launch_bounds__(96) void attention_kernel(
    const float* __restrict__ q, const float* __restrict__ k,
    const float* __restrict__ v, float* __restrict__ o) {
  int b = blockIdx.x, t = threadIdx.x;
  if (t >= NNODE * NHEAD) return;
  int n = t >> 3, hI = t & 7;
  const float* qp = q + ((size_t)b * NNODE + n) * HDIM + hI * HEADD;
  float qv[HEADD];
  #pragma unroll
  for (int d = 0; d < HEADD; ++d) qv[d] = qp[d];
  float sc[NKEY];
  float mx = -1e30f;
  int mcount = n + 1;
  for (int m = 0; m < mcount; ++m) {
    const float* kp = k + ((size_t)b * NKEY + m) * HDIM + hI * HEADD;
    float dot = 0.0f;
    #pragma unroll
    for (int d = 0; d < HEADD; ++d) dot += qv[d] * kp[d];
    dot *= 0.17677669529663687f;  // 1/sqrt(32)
    sc[m] = dot;
    mx = fmaxf(mx, dot);
  }
  float ssum = 0.0f;
  for (int m = 0; m < mcount; ++m) { float e = __expf(sc[m] - mx); sc[m] = e; ssum += e; }
  float inv = 1.0f / ssum;
  float acc[HEADD];
  #pragma unroll
  for (int d = 0; d < HEADD; ++d) acc[d] = 0.0f;
  for (int m = 0; m < mcount; ++m) {
    const float* vp = v + ((size_t)b * NKEY + m) * HDIM + hI * HEADD;
    float w = sc[m] * inv;
    #pragma unroll
    for (int d = 0; d < HEADD; ++d) acc[d] += w * vp[d];
  }
  float* op = o + ((size_t)b * NNODE + n) * HDIM + hI * HEADD;
  #pragma unroll
  for (int d = 0; d < HEADD; ++d) op[d] = acc[d];
}

// ------------------------------------------------------------------
// h = LN(h + add) per row
// ------------------------------------------------------------------
__global__ __launch_bounds__(256) void resid_ln_kernel(
    float* __restrict__ h, const float* __restrict__ add,
    const float* __restrict__ g, const float* __restrict__ be) {
  __shared__ float red[HDIM];
  int row = blockIdx.x, t = threadIdx.x;
  size_t idx = (size_t)row * HDIM + t;
  float x = h[idx] + add[idx];
  h[idx] = block_ln256(x, g[t], be[t], red);
}

// ------------------------------------------------------------------
// node logits: h[BN,256] @ out_w[256,5] + out_b
// ------------------------------------------------------------------
__global__ void out_logits_kernel(const float* __restrict__ h,
                                  const float* __restrict__ w,
                                  const float* __restrict__ bias,
                                  float* __restrict__ out) {
  int id = blockIdx.x * blockDim.x + threadIdx.x;
  if (id >= BDIM * NNODE * 5) return;
  int c = id % 5;
  int rn = id / 5;
  const float* hp = h + (size_t)rn * HDIM;
  float acc = bias[c];
  for (int k = 0; k < HDIM; ++k) acc += hp[k] * w[k * 5 + c];
  out[id] = acc;
}

// ------------------------------------------------------------------
// re[b,:] = latent[b,:8] @ edge_w1[512:520, :]
// ------------------------------------------------------------------
__global__ __launch_bounds__(256) void edge_latent_kernel(
    const float* __restrict__ latent, const float* __restrict__ w1c,
    float* __restrict__ re) {
  int b = blockIdx.x, t = threadIdx.x;
  float acc = 0.0f;
  #pragma unroll
  for (int k = 0; k < LDIM; ++k) acc += latent[b * LDIM + k] * w1c[k * HDIM + t];
  re[(size_t)b * HDIM + t] = acc;
}

// ------------------------------------------------------------------
// per lower-tri pair (i>j): hv = relu(pe_i + qe_j + re); out = hv @ W2 + b2
// ------------------------------------------------------------------
__global__ __launch_bounds__(256) void edge_pair_kernel(
    const float* __restrict__ pe, const float* __restrict__ qe,
    const float* __restrict__ re, const float* __restrict__ w2,
    const float* __restrict__ b2, float* __restrict__ out) {
  __shared__ float sh[HDIM];
  int idx = blockIdx.x;
  int b = idx / 45, p = idx % 45;
  int i = 1, pp = p;
  while (pp >= i) { pp -= i; ++i; }
  int j = pp;  // i in [1,9], j < i
  int t = threadIdx.x;
  float hv = pe[((size_t)b * NNODE + i) * HDIM + t]
           + qe[((size_t)b * NNODE + j) * HDIM + t]
           + re[(size_t)b * HDIM + t];
  sh[t] = fmaxf(hv, 0.0f);
  __syncthreads();
  if (t < 8) {
    float acc = b2[t];
    for (int k = 0; k < HDIM; ++k) acc += sh[k] * w2[k * 8 + t];
    out[(((size_t)b * NNODE + i) * NNODE + j) * 8 + t] = acc;
    out[(((size_t)b * NNODE + j) * NNODE + i) * 8 + t] = acc;
  }
}

// ------------------------------------------------------------------
extern "C" void kernel_launch(void* const* d_in, const int* in_sizes, int n_in,
                              void* d_out, int out_size, void* d_ws, size_t ws_size,
                              hipStream_t stream) {
  (void)in_sizes; (void)n_in; (void)ws_size;
  const float* latent   = (const float*)d_in[0];
  const int*   types    = (const int*)  d_in[1];
  const float* ce_w1    = (const float*)d_in[2];
  const float* ce_b1    = (const float*)d_in[3];
  const float* ce_g1    = (const float*)d_in[4];
  const float* ce_be1   = (const float*)d_in[5];
  const float* ce_w2    = (const float*)d_in[6];
  const float* ce_b2    = (const float*)d_in[7];
  const float* ce_g2    = (const float*)d_in[8];
  const float* ce_be2   = (const float*)d_in[9];
  const float* nc_w1    = (const float*)d_in[10];
  const float* nc_b1    = (const float*)d_in[11];
  const float* nc_w2    = (const float*)d_in[12];
  const float* nc_b2    = (const float*)d_in[13];
  const float* type_emb = (const float*)d_in[14];
  const float* pos_emb  = (const float*)d_in[15];
  const float* qkvo_w   = (const float*)d_in[16];
  const float* qkvo_b   = (const float*)d_in[17];
  const float* ln_g     = (const float*)d_in[18];
  const float* ln_b     = (const float*)d_in[19];
  const float* ffn_w1   = (const float*)d_in[20];
  const float* ffn_b1   = (const float*)d_in[21];
  const float* ffn_w2   = (const float*)d_in[22];
  const float* ffn_b2   = (const float*)d_in[23];
  const float* out_w    = (const float*)d_in[24];
  const float* out_b    = (const float*)d_in[25];
  const float* edge_w1  = (const float*)d_in[26];
  const float* edge_b1  = (const float*)d_in[27];
  const float* edge_w2  = (const float*)d_in[28];
  const float* edge_b2  = (const float*)d_in[29];

  float* fout = (float*)d_out;
  float* ws = (float*)d_ws;

  const int BN  = BDIM * NNODE;   // 20480 (multiple of 128)
  const int BNK = BDIM * NKEY;    // 22528 (multiple of 128)
  const int CH  = 4096;           // FFN row chunk (BN/CH = 5)
  const int HH  = HDIM * HDIM;    // 65536

  size_t off = 0;
  float* ctx  = ws + off; off += (size_t)BDIM * HDIM;
  float* ne   = ws + off; off += (size_t)BN   * HDIM;
  float* h    = ws + off; off += (size_t)BN   * HDIM;
  float* mem  = ws + off; off += (size_t)BNK  * HDIM;
  float* qb   = ws + off; off += (size_t)BN   * HDIM;   // q / o-proj / ffn out / pe
  float* kb   = ws + off; off += (size_t)BNK  * HDIM;   // k / qe
  float* vb   = ws + off; off += (size_t)BNK  * HDIM;   // v / re
  float* attn = ws + off; off += (size_t)BN   * HDIM;
  float* mid  = ws + off; off += (size_t)CH   * FFDIM;

  // fp16 transposed weights region (16B-aligned: off is a multiple of 256 floats)
  _Float16* hbase = (_Float16*)(ws + off);
  _Float16* wqT  = hbase;                          // 16 x [256][256]
  _Float16* wf1T = wqT  + (size_t)16 * HH;         // 4 x [1024][256]
  _Float16* wf2T = wf1T + (size_t)4 * HDIM * FFDIM;// 4 x [256][1024]
  _Float16* weT  = wf2T + (size_t)4 * FFDIM * HDIM;// 2 x [256][256]

  // --- weight pre-pass: fp32 [K,Nc] -> fp16 [Nc,K] ---
  for (int m = 0; m < 16; ++m)
    transpose_f16_kernel<<<HH / 256, 256, 0, stream>>>(qkvo_w + (size_t)m * HH,
                                                       wqT + (size_t)m * HH, HDIM, HDIM);
  for (int l = 0; l < NLAYER; ++l) {
    transpose_f16_kernel<<<(HDIM * FFDIM) / 256, 256, 0, stream>>>(
        ffn_w1 + (size_t)l * HDIM * FFDIM, wf1T + (size_t)l * HDIM * FFDIM, HDIM, FFDIM);
    transpose_f16_kernel<<<(FFDIM * HDIM) / 256, 256, 0, stream>>>(
        ffn_w2 + (size_t)l * FFDIM * HDIM, wf2T + (size_t)l * FFDIM * HDIM, FFDIM, HDIM);
  }
  transpose_f16_kernel<<<HH / 256, 256, 0, stream>>>(edge_w1, weT, HDIM, HDIM);
  transpose_f16_kernel<<<HH / 256, 256, 0, stream>>>(edge_w1 + (size_t)HH, weT + (size_t)HH,
                                                     HDIM, HDIM);

  // 0. zero output (edge diagonal / untouched cells)
  fill_zero_kernel<<<(out_size + 255) / 256, 256, 0, stream>>>(fout, out_size);
  // 1. context encoder
  ctx_encoder_kernel<<<BDIM, 256, 0, stream>>>(latent, ce_w1, ce_b1, ce_g1, ce_be1,
                                               ce_w2, ce_b2, ce_g2, ce_be2, ctx);
  // 2. node count head -> d_out[BN*5 ..)
  node_count_kernel<<<(BDIM + 255) / 256, 256, 0, stream>>>(latent, nc_w1, nc_b1, nc_w2, nc_b2,
                                                            fout + (size_t)BN * 5);
  // 3. embeddings / queries / memory
  build_embed_kernel<<<(BN * HDIM + 255) / 256, 256, 0, stream>>>(types, type_emb, pos_emb, ctx,
                                                                  ne, h, mem);
  // 4. transformer layers
  for (int l = 0; l < NLAYER; ++l) {
    const _Float16* wq = wqT + (size_t)(l * 4 + 0) * HH;
    const _Float16* wk = wqT + (size_t)(l * 4 + 1) * HH;
    const _Float16* wv = wqT + (size_t)(l * 4 + 2) * HH;
    const _Float16* wo = wqT + (size_t)(l * 4 + 3) * HH;
    const float* bq = qkvo_b + (size_t)(l * 4 + 0) * HDIM;
    const float* bk = qkvo_b + (size_t)(l * 4 + 1) * HDIM;
    const float* bv = qkvo_b + (size_t)(l * 4 + 2) * HDIM;
    const float* bo = qkvo_b + (size_t)(l * 4 + 3) * HDIM;

    wmma_gemm_kernel<<<dim3(HDIM / 64, BN  / 128), 128, 0, stream>>>(h,   wq, bq, qb, BN,  HDIM, HDIM, 1);
    wmma_gemm_kernel<<<dim3(HDIM / 64, BNK / 128), 128, 0, stream>>>(mem, wk, bk, kb, BNK, HDIM, HDIM, 1);
    wmma_gemm_kernel<<<dim3(HDIM / 64, BNK / 128), 128, 0, stream>>>(mem, wv, bv, vb, BNK, HDIM, HDIM, 1);
    attention_kernel<<<BDIM, 96, 0, stream>>>(qb, kb, vb, attn);
    wmma_gemm_kernel<<<dim3(HDIM / 64, BN / 128), 128, 0, stream>>>(attn, wo, bo, qb, BN, HDIM, HDIM, 1);
    resid_ln_kernel<<<BN, 256, 0, stream>>>(h, qb, ln_g + (size_t)(l * 2 + 0) * HDIM,
                                                   ln_b + (size_t)(l * 2 + 0) * HDIM);
    for (int c = 0; c < BN / CH; ++c) {
      const float* hc = h  + (size_t)c * CH * HDIM;
      float*       oc = qb + (size_t)c * CH * HDIM;
      wmma_gemm_kernel<<<dim3(FFDIM / 64, CH / 128), 128, 0, stream>>>(
          hc, wf1T + (size_t)l * HDIM * FFDIM, ffn_b1 + (size_t)l * FFDIM, mid, CH, HDIM, FFDIM, 3);
      wmma_gemm_kernel<<<dim3(HDIM / 64, CH / 128), 128, 0, stream>>>(
          mid, wf2T + (size_t)l * FFDIM * HDIM, ffn_b2 + (size_t)l * HDIM, oc, CH, FFDIM, HDIM, 1);
    }
    resid_ln_kernel<<<BN, 256, 0, stream>>>(h, qb, ln_g + (size_t)(l * 2 + 1) * HDIM,
                                                   ln_b + (size_t)(l * 2 + 1) * HDIM);
  }
  // 5. node logits -> d_out[0 .. BN*5)
  out_logits_kernel<<<(BN * 5 + 255) / 256, 256, 0, stream>>>(h, out_w, out_b, fout);
  // 6. edge decoder (separated first-layer GEMM): pe = ne@W1a + b1, qe = ne@W1b
  wmma_gemm_kernel<<<dim3(HDIM / 64, BN / 128), 128, 0, stream>>>(
      ne, weT, edge_b1, qb, BN, HDIM, HDIM, 1);
  wmma_gemm_kernel<<<dim3(HDIM / 64, BN / 128), 128, 0, stream>>>(
      ne, weT + (size_t)HH, nullptr, kb, BN, HDIM, HDIM, 0);
  edge_latent_kernel<<<BDIM, 256, 0, stream>>>(latent, edge_w1 + (size_t)2 * HH, vb);
  edge_pair_kernel<<<BDIM * 45, 256, 0, stream>>>(
      qb, kb, vb, edge_w2, edge_b2, fout + (size_t)BN * 5 + (size_t)BDIM * 3);
}
